// SphericalDecoder_53532472377799
// MI455X (gfx1250) — compile-verified, matrix operand order checked by reading, not run on hardware
//
#include <hip/hip_runtime.h>
#include <hip/hip_bf16.h>
#include <math.h>

typedef __attribute__((ext_vector_type(16))) _Float16 v16h;
typedef __attribute__((ext_vector_type(8)))  float    v8f;
typedef __attribute__((ext_vector_type(4)))  float    v4f;

// ---- CDNA5 async global->LDS copy (ASYNCcnt-tracked, ISA 15.18.3 op 98) ----
__device__ __forceinline__ void async_b128(uint32_t ldsOff, const void* gaddr)
{
    asm volatile("global_load_async_to_lds_b128 %0, %1, off"
                 :: "v"(ldsOff), "v"(gaddr) : "memory");
}
template<int N>
__device__ __forceinline__ void wait_async_le()
{
    asm volatile("s_wait_asynccnt %0" :: "n"(N) : "memory");
}

// ---------------------------------------------------------------------------
// Fused gather + GEMM + epilogue using v_wmma_f32_16x16x32_f16.
// D[m, nOff+n] = sum_k A[m,k] * W[n,k] (+ bias + addB[b,n] + addM[m,n])
// A[m,k]: tap = k / actC, ch = k % actC, src row = b*V + neigh[v*9+tap]
//         (or m itself when neigh == nullptr). fp32 in memory, cvt->f16.
// B: pre-packed f16 fragments; staged per-workgroup into LDS in 128-K
//    super-chunks via global_load_async_to_lds_b128, double buffered.
// A fragment layout per CDNA5 ISA (16-bit A 16x32):
//   lanes 0-15: M=lane, K in {0..7,16..23}; lanes 16-31: +8 on each K octet.
// C/D layout: VGPR r -> row mBase + r + 8*(lane>>4), col = lane&15.
// Requirements satisfied by all call sites: M%128==0, K%128==0, actC%32==0.
// ---------------------------------------------------------------------------
template<int NT>
__global__ __launch_bounds__(256) void gemm_gather_wmma(
    const float* __restrict__ act, const int* __restrict__ neigh,
    const _Float16* __restrict__ Bp, const float* __restrict__ bias,
    const float* __restrict__ addB, const float* __restrict__ addM,
    float* __restrict__ out, int M, int K, int actC, int V, int Nout, int nOff)
{
    // 2 buffers * NT n-tiles * 4 kt * 32 lanes * 16 halves  (NT=8 -> 64 KB)
    __shared__ __align__(32) _Float16 ldsb[2 * NT * 2048];

    const int tid   = threadIdx.x;
    const int lane  = tid & 31;
    const int wave  = tid >> 5;
    const int mBase = ((int)blockIdx.x * 8 + wave) * 16;   // grid is exact
    const int half_ = lane >> 4;
    const int mrow  = mBase + (lane & 15);
    const int b     = mBase / V;              // V % 16 == 0 -> uniform per tile
    const int v     = mrow - b * V;
    const int KT    = K >> 5;
    const int NSC   = KT >> 2;                // 128-K super-chunks
    const int cChunks = actC >> 5;            // % 4 == 0 at every call site
    const int taps  = K / actC;               // 9 for gathers, 1 for plain GEMM

    const uint32_t ldsBase = (uint32_t)(uintptr_t)(&ldsb[0]);

    auto issue_copy = [&](int sc, uint32_t bufSel) {
#pragma unroll
        for (int i = 0; i < NT; ++i) {
            int l     = i * 256 + tid;        // packed 16B chunk id
            int half8 = l & 1;
            int lane_ = (l >> 1) & 31;
            int ktL   = (l >> 6) & 3;
            int nt    = l >> 8;
            size_t srcHalf = ((size_t)nt * KT + (size_t)(sc * 4 + ktL)) * 512
                             + (size_t)lane_ * 16 + (size_t)half8 * 8;
            async_b128(ldsBase + bufSel * (uint32_t)(NT * 4096)
                               + (uint32_t)l * 16u,
                       (const void*)(Bp + srcHalf));
        }
    };

    v8f zero;
#pragma unroll
    for (int r = 0; r < 8; ++r) zero[r] = 0.f;
    v8f acc[NT];
#pragma unroll
    for (int t = 0; t < NT; ++t) acc[t] = zero;

    issue_copy(0, 0);                          // prologue: stage super-chunk 0

    int kt = 0;
    for (int tap = 0; tap < taps; ++tap) {
        size_t srow = neigh ? ((size_t)b * V + (size_t)neigh[(size_t)v * 9 + tap])
                            : (size_t)mrow;
        const float* __restrict__ row = act + srow * actC;
        for (int cc = 0; cc < cChunks; ++cc, ++kt) {
            if ((kt & 3) == 0) {               // super-chunk boundary (uniform)
                int sc = kt >> 2;
                if (sc > 0) __syncthreads();   // everyone done with buf[(sc+1)&1]
                if (sc + 1 < NSC) {
                    issue_copy(sc + 1, (uint32_t)((sc + 1) & 1));
                    wait_async_le<NT>();       // batch sc complete (in-order)
                } else {
                    wait_async_le<0>();
                }
                __syncthreads();               // buf[sc&1] visible to all waves
            }
            // ---- A fragment: gather 32 K of fp32, convert to f16 ----
            const float* p0 = row + cc * 32 + half_ * 8;   // 32B aligned
            v4f f0 = *(const v4f*)(p0);
            v4f f1 = *(const v4f*)(p0 + 4);
            v4f f2 = *(const v4f*)(p0 + 16);
            v4f f3 = *(const v4f*)(p0 + 20);
            v16h a;
#pragma unroll
            for (int j = 0; j < 4; ++j) {
                a[j]      = (_Float16)f0[j];
                a[4 + j]  = (_Float16)f1[j];
                a[8 + j]  = (_Float16)f2[j];
                a[12 + j] = (_Float16)f3[j];
            }
            // ---- B fragments from LDS ----
            const _Float16* bbase = ldsb + (size_t)((kt >> 2) & 1) * (NT * 2048)
                                  + (size_t)((kt & 3) * 32 + lane) * 16;
#pragma unroll
            for (int t = 0; t < NT; ++t) {
                v16h bf = *(const v16h*)(bbase + (size_t)t * 2048);
                acc[t] = __builtin_amdgcn_wmma_f32_16x16x32_f16(
                    false, a, false, bf, (short)0, acc[t], false, false);
            }
        }
    }

    const int col   = lane & 15;
    const int rbase = mBase + half_ * 8;
#pragma unroll
    for (int t = 0; t < NT; ++t) {
        const int n = nOff + t * 16 + col;
        const bool nOK = (n < Nout);
        const float bv = (bias && nOK) ? bias[n] : 0.f;
        const float tv = (addB && nOK) ? addB[(size_t)b * Nout + n] : 0.f;
#pragma unroll
        for (int r = 0; r < 8; ++r) {
            const int rowi = rbase + r;
            float val = acc[t][r] + bv + tv;
            if (nOK) {
                size_t oidx = (size_t)rowi * Nout + n;
                if (addM) val += addM[oidx];
                out[oidx] = val;
            }
        }
    }
}

// Pack fp32 weight (O x K row-major; or conv OIHW with convCin) into the f16
// B-fragment layout consumed above. One thread per packed half.
__global__ void pack_w_kernel(const float* __restrict__ W, _Float16* __restrict__ Bp,
                              int K, int O, int convCin, int KT, size_t total)
{
    size_t idx = (size_t)blockIdx.x * blockDim.x + threadIdx.x;
    if (idx >= total) return;
    int j     = (int)(idx & 15);
    int lane  = (int)((idx >> 4) & 31);
    size_t rest = idx >> 9;
    int kt = (int)(rest % (size_t)KT);
    int nt = (int)(rest / (size_t)KT);
    int half_ = lane >> 4;
    int n = nt * 16 + (lane & 15);
    int kLocal = (j & 7) + ((j >> 3) << 4) + (half_ << 3);
    int k = kt * 32 + kLocal;
    float val = 0.f;
    if (n < O && k < K) {
        if (convCin) {  // conv weight (O, Cin, 3, 3); our k = tap*Cin + ch
            int tap = k / convCin;
            int ch  = k - tap * convCin;
            val = W[((size_t)n * convCin + ch) * 9 + tap];
        } else {
            val = W[(size_t)n * K + k];
        }
    }
    Bp[idx] = (_Float16)val;
}

// Circular-pad 3x3 neighbor table for an H x W grid (tap = (dh+1)*3 + dw+1).
__global__ void build_wrap_neigh_kernel(int* __restrict__ neigh, int H, int W)
{
    int v = blockIdx.x * blockDim.x + threadIdx.x;
    if (v >= H * W) return;
    int h = v / W, w = v - h * W;
#pragma unroll
    for (int tap = 0; tap < 9; ++tap) {
        int dh = tap / 3 - 1, dw = tap % 3 - 1;
        int hh = h + dh; hh = (hh < 0) ? hh + H : (hh >= H ? hh - H : hh);
        int ww = w + dw; ww = (ww < 0) ? ww + W : (ww >= W ? ww - W : ww);
        neigh[v * 9 + tap] = hh * W + ww;
    }
}

// GroupNorm stats: one block per (batch, group); population mean/var over V*cg.
__global__ __launch_bounds__(256) void gn_stats_kernel(
    const float* __restrict__ x, float* __restrict__ stats, int V, int C)
{
    const int bg = blockIdx.x, b = bg >> 5, g = bg & 31;
    const int cg = C >> 5;
    float s = 0.f, s2 = 0.f;
    for (int v = threadIdx.x; v < V; v += blockDim.x) {
        const float* p = x + ((size_t)b * V + v) * C + g * cg;
        for (int c = 0; c < cg; ++c) { float t = p[c]; s += t; s2 += t * t; }
    }
    __shared__ float sh1[256], sh2[256];
    sh1[threadIdx.x] = s; sh2[threadIdx.x] = s2;
    __syncthreads();
    for (int st = 128; st > 0; st >>= 1) {
        if ((int)threadIdx.x < st) {
            sh1[threadIdx.x] += sh1[threadIdx.x + st];
            sh2[threadIdx.x] += sh2[threadIdx.x + st];
        }
        __syncthreads();
    }
    if (threadIdx.x == 0) {
        float n  = (float)V * (float)cg;
        float mu = sh1[0] / n;
        float var = sh2[0] / n - mu * mu;
        stats[bg * 2]     = mu;
        stats[bg * 2 + 1] = rsqrtf(var + 1e-5f);
    }
}

// Fused normalize * gamma + beta, then SiLU.
__global__ void gn_apply_silu_kernel(const float* __restrict__ x,
                                     const float* __restrict__ stats,
                                     const float* __restrict__ gamma,
                                     const float* __restrict__ beta,
                                     float* __restrict__ out,
                                     int V, int C, size_t total)
{
    size_t idx = (size_t)blockIdx.x * blockDim.x + threadIdx.x;
    if (idx >= total) return;
    int c = (int)(idx % (size_t)C);
    size_t rowv = idx / (size_t)C;
    int b = (int)(rowv / (size_t)V);
    int g = c / (C >> 5);
    float mu = stats[(b * 32 + g) * 2];
    float rs = stats[(b * 32 + g) * 2 + 1];
    float y = (x[idx] - mu) * rs * gamma[c] + beta[c];
    out[idx] = y * (1.f / (1.f + __expf(-y)));
}

// Bilinear 2x upsample, half-pixel centers with clamping (matches reference).
__global__ void bilinear_up2_kernel(const float* __restrict__ x, float* __restrict__ y,
                                    int B, int H, int W, int C, size_t total)
{
    size_t idx = (size_t)blockIdx.x * blockDim.x + threadIdx.x;
    if (idx >= total) return;
    int c = (int)(idx % (size_t)C);
    size_t r = idx / (size_t)C;
    int ow = (int)(r % (size_t)(2 * W)); r /= (size_t)(2 * W);
    int oh = (int)(r % (size_t)(2 * H));
    int b  = (int)(r / (size_t)(2 * H));
    float sr = fminf(fmaxf(oh * 0.5f - 0.25f, 0.f), (float)(H - 1));
    int r0 = (int)floorf(sr); int r1 = min(r0 + 1, H - 1); float wr = sr - (float)r0;
    float sc = fminf(fmaxf(ow * 0.5f - 0.25f, 0.f), (float)(W - 1));
    int c0 = (int)floorf(sc); int c1 = min(c0 + 1, W - 1); float wc = sc - (float)c0;
    const float* base = x + (size_t)b * H * W * C + c;
    float v00 = base[((size_t)r0 * W + c0) * C];
    float v01 = base[((size_t)r0 * W + c1) * C];
    float v10 = base[((size_t)r1 * W + c0) * C];
    float v11 = base[((size_t)r1 * W + c1) * C];
    float top = v00 * (1.f - wc) + v01 * wc;
    float bot = v10 * (1.f - wc) + v11 * wc;
    y[idx] = top * (1.f - wr) + bot * wr;
}

// t = silu(t_emb) @ t_w.T + t_b   (tiny: B x 128)
__global__ void time_vec_kernel(const float* __restrict__ t_emb,
                                const float* __restrict__ tw,
                                const float* __restrict__ tb,
                                float* __restrict__ out, int B, int N)
{
    int i = blockIdx.x * blockDim.x + threadIdx.x;
    if (i >= B * N) return;
    int b = i / N, o = i - b * N;
    float s = tb[o];
#pragma unroll
    for (int j = 0; j < 16; ++j) {
        float e = t_emb[b * 16 + j];
        e = e * (1.f / (1.f + __expf(-e)));
        s += e * tw[o * 16 + j];
    }
    out[i] = s;
}

// out[r, 0:Ca] = a[r], out[r, Ca:Ca+Cb] = bsrc[r]
__global__ void concat_kernel(const float* __restrict__ a, int Ca,
                              const float* __restrict__ bsrc, int Cb,
                              float* __restrict__ out, size_t total)
{
    size_t idx = (size_t)blockIdx.x * blockDim.x + threadIdx.x;
    if (idx >= total) return;
    int C = Ca + Cb;
    int c = (int)(idx % (size_t)C);
    size_t r = idx / (size_t)C;
    out[idx] = (c < Ca) ? a[r * (size_t)Ca + c] : bsrc[r * (size_t)Cb + (c - Ca)];
}

// ---------------------------------------------------------------------------
extern "C" void kernel_launch(void* const* d_in, const int* in_sizes, int n_in,
                              void* d_out, int out_size, void* d_ws, size_t ws_size,
                              hipStream_t stream)
{
    (void)in_sizes; (void)n_in; (void)out_size; (void)ws_size;
    const int B = 4, V1 = 64 * 128, V2 = 128 * 256;
    const int M1 = B * V1, M2 = B * V2;

    // ---- inputs (setup_inputs dict order, params flattened in insertion order)
    const float* enc0  = (const float*)d_in[0];
    const float* enc1  = (const float*)d_in[1];
    const float* enc2  = (const float*)d_in[2];
    const float* t_emb = (const float*)d_in[3];
    const int*   ngh1  = (const int*)d_in[4];
    const int*   ngh2  = (const int*)d_in[5];
    auto F = [&](int i) { return (const float*)d_in[i]; };
    // params: 6..9 up0{w,b,g,beta}; 10..21 l0b0 (with sc); 22..31 l0b1;
    //         32..35 up1; 36..47 l1b0 (with sc); 48..57 l1b1; 58,59 fn; 60,61 f

    // ---- workspace carve-out (~360 MB; 256B aligned slots)
    char* wsb = (char*)d_ws;
    size_t off = 0;
    auto alloc = [&](size_t bytes) -> void* {
        off = (off + 255) & ~(size_t)255;
        void* p = wsb + off;
        off += bytes;
        return p;
    };
    auto allocH = [&](size_t halves) { return (_Float16*)alloc(halves * 2); };
    auto allocF = [&](size_t n)      { return (float*)alloc(n * 4); };

    _Float16* pk_up0  = allocH((size_t)16 * 2304 * 16);   // 16 n-tiles, split use
    _Float16* pk_b0c1 = allocH((size_t)8  * 3456 * 16);
    _Float16* pk_b0c2 = allocH((size_t)8  * 1152 * 16);
    _Float16* pk_b0sc = allocH((size_t)8  * 384  * 16);
    _Float16* pk_b1c1 = allocH((size_t)8  * 1152 * 16);
    _Float16* pk_b1c2 = allocH((size_t)8  * 1152 * 16);
    _Float16* pk_up1  = allocH((size_t)8  * 1152 * 16);
    _Float16* pk_b2c1 = allocH((size_t)8  * 2304 * 16);
    _Float16* pk_b2c2 = allocH((size_t)8  * 1152 * 16);
    _Float16* pk_b2sc = allocH((size_t)8  * 256  * 16);
    _Float16* pk_b3c1 = allocH((size_t)8  * 1152 * 16);
    _Float16* pk_b3c2 = allocH((size_t)8  * 1152 * 16);
    _Float16* pk_f    = allocH((size_t)1  * 1152 * 16);
    int*   cn1   = (int*)alloc((size_t)V1 * 9 * 4);
    int*   cn2   = (int*)alloc((size_t)V2 * 9 * 4);
    float* stats = allocF(B * 32 * 2);
    float* tvec  = allocF(B * 128);
    const size_t NA = (size_t)M2 * 128;        // 16.78M floats
    float* SLA = allocF(NA);
    float* SLB = allocF(NA);
    float* SLC = allocF((size_t)M2 * 256);     // 33.55M floats
    float* SLD = allocF(NA);
    float* SLE = allocF((size_t)M1 * 128);     // x2 only

    // ---- helpers
    auto packW = [&](const float* src, _Float16* dst, int K, int O, int convCin, int NT) {
        size_t tot = (size_t)NT * K * 16;
        pack_w_kernel<<<(unsigned)((tot + 255) / 256), 256, 0, stream>>>(
            src, dst, K, O, convCin, K >> 5, tot);
    };
    auto gemm = [&](int NT, const float* act, const int* ng, const _Float16* bp,
                    const float* bias, const float* aB, const float* aM, float* outp,
                    int M, int K, int actC, int V, int Nout, int nOff) {
        dim3 g((unsigned)(M / 128)), blk(256);
        if (NT == 1)
            gemm_gather_wmma<1><<<g, blk, 0, stream>>>(act, ng, bp, bias, aB, aM,
                                                       outp, M, K, actC, V, Nout, nOff);
        else
            gemm_gather_wmma<8><<<g, blk, 0, stream>>>(act, ng, bp, bias, aB, aM,
                                                       outp, M, K, actC, V, Nout, nOff);
    };
    auto gnorm_silu = [&](const float* x, const float* g, const float* beta,
                          float* outp, int V, int C) {
        gn_stats_kernel<<<B * 32, 256, 0, stream>>>(x, stats, V, C);
        size_t tot = (size_t)B * V * C;
        gn_apply_silu_kernel<<<(unsigned)((tot + 255) / 256), 256, 0, stream>>>(
            x, stats, g, beta, outp, V, C, tot);
    };
    auto bilinear = [&](const float* x, float* y, int H, int W, int C) {
        size_t tot = (size_t)B * 4 * H * W * C;
        bilinear_up2_kernel<<<(unsigned)((tot + 255) / 256), 256, 0, stream>>>(
            x, y, B, H, W, C, tot);
    };
    auto concat = [&](const float* a, int Ca, const float* bsrc, int Cb, float* o, size_t rows) {
        size_t tot = rows * (size_t)(Ca + Cb);
        concat_kernel<<<(unsigned)((tot + 255) / 256), 256, 0, stream>>>(a, Ca, bsrc, Cb, o, tot);
    };
    auto timevec = [&](const float* tw, const float* tb) {
        time_vec_kernel<<<(B * 128 + 255) / 256, 256, 0, stream>>>(t_emb, tw, tb, tvec, B, 128);
    };

    // ---- one-time-per-call prep (deterministic)
    build_wrap_neigh_kernel<<<(V1 + 255) / 256, 256, 0, stream>>>(cn1, 64, 128);
    build_wrap_neigh_kernel<<<(V2 + 255) / 256, 256, 0, stream>>>(cn2, 128, 256);
    packW(F(6),  pk_up0,  2304, 256, 256, 16);   // up0 conv (OIHW)
    packW(F(12), pk_b0c1, 3456, 128, 0, 8);
    packW(F(18), pk_b0c2, 1152, 128, 0, 8);
    packW(F(20), pk_b0sc, 384,  128, 0, 8);
    packW(F(24), pk_b1c1, 1152, 128, 0, 8);
    packW(F(30), pk_b1c2, 1152, 128, 0, 8);
    packW(F(32), pk_up1,  1152, 128, 128, 8);    // up1 conv (OIHW)
    packW(F(38), pk_b2c1, 2304, 128, 0, 8);
    packW(F(44), pk_b2c2, 1152, 128, 0, 8);
    packW(F(46), pk_b2sc, 256,  128, 0, 8);
    packW(F(50), pk_b3c1, 1152, 128, 0, 8);
    packW(F(56), pk_b3c2, 1152, 128, 0, 8);
    packW(F(60), pk_f,    1152, 3,   0, 1);

    // ---- level 1 ----
    bilinear(enc0, SLC, 32, 64, 256);                                   // x0 @ SLC
    // up0 conv: 256 output channels as two 8-tile passes over same packed buf
    gemm(8, SLC, cn1, pk_up0, F(7), nullptr, nullptr, SLA, M1, 2304, 256, V1, 256, 0);
    gemm(8, SLC, cn1, pk_up0 + (size_t)8 * 72 * 512, F(7), nullptr, nullptr,
         SLA, M1, 2304, 256, V1, 256, 128);
    gnorm_silu(SLA, F(8), F(9), SLA, V1, 256);                          // up0 out
    concat(SLA, 256, enc1, 128, SLC, (size_t)M1);                       // cat1 @ SLC
    // l0b0 (384 -> 128, has shortcut)
    gemm(8, SLC, nullptr, pk_b0sc, F(21), nullptr, nullptr, SLD, M1, 384, 384, V1, 128, 0);
    gnorm_silu(SLC, F(10), F(11), SLC, V1, 384);
    timevec(F(14), F(15));
    gemm(8, SLC, ngh1, pk_b0c1, F(13), tvec, nullptr, SLB, M1, 3456, 384, V1, 128, 0);
    gnorm_silu(SLB, F(16), F(17), SLB, V1, 128);
    gemm(8, SLB, ngh1, pk_b0c2, F(19), nullptr, SLD, SLA, M1, 1152, 128, V1, 128, 0); // x1
    // l0b1 (128 -> 128)
    gnorm_silu(SLA, F(22), F(23), SLC, V1, 128);
    timevec(F(26), F(27));
    gemm(8, SLC, ngh1, pk_b1c1, F(25), tvec, nullptr, SLB, M1, 1152, 128, V1, 128, 0);
    gnorm_silu(SLB, F(28), F(29), SLB, V1, 128);
    gemm(8, SLB, ngh1, pk_b1c2, F(31), nullptr, SLA, SLE, M1, 1152, 128, V1, 128, 0); // x2

    // ---- level 2 ----
    bilinear(SLE, SLA, 64, 128, 128);                                   // u @ SLA
    gemm(8, SLA, cn2, pk_up1, F(33), nullptr, nullptr, SLB, M2, 1152, 128, V2, 128, 0);
    gnorm_silu(SLB, F(34), F(35), SLB, V2, 128);
    concat(SLB, 128, enc2, 128, SLC, (size_t)M2);                       // cat2 @ SLC
    // l1b0 (256 -> 128, has shortcut)
    gemm(8, SLC, nullptr, pk_b2sc, F(47), nullptr, nullptr, SLD, M2, 256, 256, V2, 128, 0);
    gnorm_silu(SLC, F(36), F(37), SLC, V2, 256);
    timevec(F(40), F(41));
    gemm(8, SLC, ngh2, pk_b2c1, F(39), tvec, nullptr, SLB, M2, 2304, 256, V2, 128, 0);
    gnorm_silu(SLB, F(42), F(43), SLB, V2, 128);
    gemm(8, SLB, ngh2, pk_b2c2, F(45), nullptr, SLD, SLA, M2, 1152, 128, V2, 128, 0); // x3
    // l1b1 (128 -> 128)
    gnorm_silu(SLA, F(48), F(49), SLC, V2, 128);
    timevec(F(52), F(53));
    gemm(8, SLC, ngh2, pk_b3c1, F(51), tvec, nullptr, SLB, M2, 1152, 128, V2, 128, 0);
    gnorm_silu(SLB, F(54), F(55), SLB, V2, 128);
    gemm(8, SLB, ngh2, pk_b3c2, F(57), nullptr, SLA, SLD, M2, 1152, 128, V2, 128, 0); // x4
    // ---- head ----
    gnorm_silu(SLD, F(58), F(59), SLC, V2, 128);
    gemm(1, SLC, ngh2, pk_f, F(61), nullptr, nullptr, (float*)d_out, M2, 1152, 128, V2, 3, 0);
}
